// InjectedEncoder_10479720202458
// MI455X (gfx1250) — matrix operand order repaired and checked by reading, not compiled, not run on hardware
//
#include <hip/hip_runtime.h>
#include <math.h>

// ---------------- problem constants ----------------
#define BATCH   32
#define NF      64          // n_filter
#define FHID    128         // FH = 2*n_filter
#define LATD    512
#define IMGD    64
#define HW      (IMGD*IMGD)               // 4096
#define ACTN    ((long long)BATCH*HW*NF)  // 8,388,608
#define SLOPE   0.2f
#define NCALLS  8

// dyna ks layout (per sample, total 37248 floats)
#define KS_TOTAL 37248
#define KIN_OFF  0
#define KMID_OFF 8192
#define KOUT_OFF 24576
#define KSH_OFF  32768
#define BIN_OFF  36864
#define BMID_OFF 36992
#define BOUT_OFF 37120
#define BSH_OFF  37184

typedef __attribute__((ext_vector_type(16))) _Float16 v16h;
typedef __attribute__((ext_vector_type(8)))  float    v8f;
typedef __attribute__((ext_vector_type(4)))  int      v4i;

// CDNA5 async global->LDS path (guarded; integer casts keep the guard compile-safe)
#if defined(__gfx1250__) && __has_builtin(__builtin_amdgcn_global_load_async_to_lds_b128) && __has_builtin(__builtin_amdgcn_s_wait_asynccnt)
#define HAVE_ASYNC_LDS 1
typedef __attribute__((address_space(1))) v4i as1_v4i;
typedef __attribute__((address_space(3))) v4i as3_v4i;
#else
#define HAVE_ASYNC_LDS 0
#endif

__device__ __forceinline__ float lrelu_f(float v) { return v > 0.f ? v : SLOPE * v; }

__device__ __forceinline__ v16h pack16(float4 u0, float4 u1, float4 u2, float4 u3) {
  v16h f;
  f[0]=(_Float16)u0.x; f[1]=(_Float16)u0.y; f[2]=(_Float16)u0.z; f[3]=(_Float16)u0.w;
  f[4]=(_Float16)u1.x; f[5]=(_Float16)u1.y; f[6]=(_Float16)u1.z; f[7]=(_Float16)u1.w;
  f[8]=(_Float16)u2.x; f[9]=(_Float16)u2.y; f[10]=(_Float16)u2.z; f[11]=(_Float16)u2.w;
  f[12]=(_Float16)u3.x; f[13]=(_Float16)u3.y; f[14]=(_Float16)u3.z; f[15]=(_Float16)u3.w;
  return f;
}

// A fragment (16x32 f16, MxK): lane half h holds K = {k0+16h..+15} in e0..15
// (row = lane&15).
__device__ __forceinline__ v16h load_a_frag(const float* __restrict__ row, int ka) {
  const float4* q0 = reinterpret_cast<const float4*>(row + ka);
  return pack16(q0[0], q0[1], q0[2], q0[3]);
}
// B fragment (32x16 f16, KxN): lane half h holds K = {k0+16h..+15} in e0..15
// (col = lane&15). W is row-major (N x K), i.e. B[k][n] = W[n][k].
__device__ __forceinline__ v16h load_b_frag(const float* __restrict__ row, int kb) {
  const float4* q = reinterpret_cast<const float4*>(row + kb);
  return pack16(q[0], q[1], q[2], q[3]);
}

#define FLAG_LRELU 1
#define FLAG_ACC   2

// ---------------- generic WMMA linear kernel (small-M path) ----------------
// C[b] (+)= act( A[b] (M x K) @ W[b] (N x K)^T + bias[b] )
// grid: x = N/16/4, y = M/16, z = batch. block = (32,4): waves split N tiles.
__global__ void wmma_lin_kernel(const float* __restrict__ A, long long aB, int lda,
                                const float* __restrict__ W, long long wB, int ldw,
                                const float* __restrict__ bias, long long bB,
                                float* __restrict__ C, long long cB, int ldc,
                                int N, int K, int flags)
{
  int ntile = blockIdx.x * blockDim.y + threadIdx.y;      // wave-uniform
  if (ntile * 16 >= N) return;                            // whole wave exits: EXEC stays all-ones
  int mtile = blockIdx.y;
  int b     = blockIdx.z;
  int lane  = threadIdx.x;
  int half  = lane >> 4;
  int r     = lane & 15;

  const float* arow = A + (long long)b * aB + (long long)(mtile * 16 + r) * lda;
  const float* wrow = W + (long long)b * wB + (long long)(ntile * 16 + r) * ldw;

  v8f acc = {0.f,0.f,0.f,0.f,0.f,0.f,0.f,0.f};
  for (int k0 = 0; k0 < K; k0 += 32) {
    v16h af = load_a_frag(arow, k0 + half * 16);
    v16h bf = load_b_frag(wrow, k0 + half * 16);
    acc = __builtin_amdgcn_wmma_f32_16x16x32_f16(false, af, false, bf,
                                                 (short)0, acc, false, false);
  }
  int col = ntile * 16 + r;
  float bv = bias ? bias[(long long)b * bB + col] : 0.f;
  float* cp = C + (long long)b * cB + (long long)(mtile * 16 + half * 8) * ldc + col;
#pragma unroll
  for (int e = 0; e < 8; ++e) {
    float v = acc[e] + bv;
    if (flags & FLAG_LRELU) v = lrelu_f(v);
    if (flags & FLAG_ACC)   v += cp[(long long)e * ldc];
    cp[(long long)e * ldc] = v;
  }
}

// ---------------- big-M WMMA kernel: LDS-staged B, 4x M-register-blocking ----
// Each block: 4 waves share one N-tile; wave computes a 64x16 C strip
// (4 WMMAs per k-step off one LDS-staged B fragment). B tile (16x32 f32, 2KB)
// is double-buffered in LDS via async global->LDS loads when available.
__device__ __forceinline__ void stage_b_tile(const float* __restrict__ wbase, int ldw,
                                             int k0, float4* lbuf, int tid)
{
  int row = tid >> 3, seg = tid & 7;                     // 16 rows x 8 float4 segs
  const float* src = wbase + (long long)row * ldw + k0 + seg * 4;
#if HAVE_ASYNC_LDS
  as1_v4i* g = (as1_v4i*)(unsigned long long)(const void*)src;
  as3_v4i* l = (as3_v4i*)(unsigned int)(unsigned long long)(void*)(lbuf + row * 8 + seg);
  __builtin_amdgcn_global_load_async_to_lds_b128(g, l, 0, 0);
#else
  lbuf[row * 8 + seg] = *reinterpret_cast<const float4*>(src);
#endif
}

__device__ __forceinline__ void wait_async_lds() {
#if HAVE_ASYNC_LDS
  __builtin_amdgcn_s_wait_asynccnt(0);
#endif
}

__global__ void __launch_bounds__(128)
wmma_lin_big_kernel(const float* __restrict__ A, long long aB, int lda,
                    const float* __restrict__ W, long long wB, int ldw,
                    const float* __restrict__ bias, long long bB,
                    float* __restrict__ C, long long cB, int ldc,
                    int N, int K, int flags)
{
  __shared__ float4 lbuf[2][128];                        // 2 x (16 rows x 32 floats)
  int wv   = threadIdx.y;                                // 0..3
  int lane = threadIdx.x;                                // 0..31
  int tid  = wv * 32 + lane;
  int half = lane >> 4;
  int r    = lane & 15;
  int ntile = blockIdx.x;
  int b     = blockIdx.z;

  const float* Ab    = A + (long long)b * aB;
  const float* wbase = W + (long long)b * wB + (long long)(ntile * 16) * ldw;

  long long mrow0 = (long long)blockIdx.y * 256 + wv * 64 + r;
  const float* ar0 = Ab + (mrow0 +  0) * lda;
  const float* ar1 = Ab + (mrow0 + 16) * lda;
  const float* ar2 = Ab + (mrow0 + 32) * lda;
  const float* ar3 = Ab + (mrow0 + 48) * lda;

  v8f acc0 = {0.f,0.f,0.f,0.f,0.f,0.f,0.f,0.f};
  v8f acc1 = acc0, acc2 = acc0, acc3 = acc0;

  stage_b_tile(wbase, ldw, 0, &lbuf[0][0], tid);

  int nsteps = K >> 5;
  for (int i = 0; i < nsteps; ++i) {
    int k0 = i << 5;
    wait_async_lds();
    __syncthreads();
    if (i + 1 < nsteps) {
      stage_b_tile(wbase, ldw, k0 + 32, &lbuf[(i + 1) & 1][0], tid);  // overlap next tile
      __builtin_prefetch(ar0 + k0 + 32, 0, 1);                        // global_prefetch_b8
      __builtin_prefetch(ar2 + k0 + 32, 0, 1);
    }
    const float4* lb = &lbuf[i & 1][r * 8 + half * 4];
    v16h bf = pack16(lb[0], lb[1], lb[2], lb[3]);
    int ka = k0 + half * 16;
    v16h a0 = load_a_frag(ar0, ka);
    v16h a1 = load_a_frag(ar1, ka);
    v16h a2 = load_a_frag(ar2, ka);
    v16h a3 = load_a_frag(ar3, ka);
    acc0 = __builtin_amdgcn_wmma_f32_16x16x32_f16(false, a0, false, bf, (short)0, acc0, false, false);
    acc1 = __builtin_amdgcn_wmma_f32_16x16x32_f16(false, a1, false, bf, (short)0, acc1, false, false);
    acc2 = __builtin_amdgcn_wmma_f32_16x16x32_f16(false, a2, false, bf, (short)0, acc2, false, false);
    acc3 = __builtin_amdgcn_wmma_f32_16x16x32_f16(false, a3, false, bf, (short)0, acc3, false, false);
  }

  int col = ntile * 16 + r;
  float bv = bias ? bias[(long long)b * bB + col] : 0.f;
  long long mbase = (long long)blockIdx.y * 256 + wv * 64 + half * 8;
  float* cb = C + (long long)b * cB + col;
  v8f accs[4] = {acc0, acc1, acc2, acc3};
#pragma unroll
  for (int t = 0; t < 4; ++t) {
#pragma unroll
    for (int e = 0; e < 8; ++e) {
      float v = accs[t][e] + bv;
      if (flags & FLAG_LRELU) v = lrelu_f(v);
      float* cp = cb + (mbase + t * 16 + e) * (long long)ldc;
      if (flags & FLAG_ACC) v += *cp;
      *cp = v;
    }
  }
}

// ---------------- input resblock (C=3, scalar) ----------------
__global__ void in_conv_kernel(const float* __restrict__ x,
                               const float* __restrict__ w0, const float* __restrict__ b0,
                               const float* __restrict__ w1, const float* __restrict__ b1,
                               const float* __restrict__ ws, const float* __restrict__ bs,
                               float* __restrict__ out, float* __restrict__ emb)
{
  int t  = threadIdx.x;                       // 0..63 : hidden idx / out channel
  int ty = threadIdx.y;
  long long pix = (long long)blockIdx.x * blockDim.y + ty;  // 0 .. BATCH*HW-1
  int b = (int)(pix / HW), p = (int)(pix % HW);

  __shared__ float sx[4][3];
  __shared__ float sh[4][NF];
  if (t < 3) sx[ty][t] = x[((long long)b * 3 + t) * HW + p];
  __syncthreads();
  float x0 = sx[ty][0], x1 = sx[ty][1], x2 = sx[ty][2];
  float h = w0[t*3+0]*x0 + w0[t*3+1]*x1 + w0[t*3+2]*x2 + b0[t];
  sh[ty][t] = lrelu_f(h);
  __syncthreads();
  float a = ws[t*3+0]*x0 + ws[t*3+1]*x1 + ws[t*3+2]*x2 + bs[t];
  for (int k = 0; k < NF; ++k) a += w1[t*NF + k] * sh[ty][k];
  out[pix * NF + t] = a;
  emb[((long long)b * NF + t) * HW + p] = a;
}

// ---------------- frac-sobel pre: h3(NCHW,3) = pre_w(3x64) @ out + pre_b ----
__global__ void sob_pre_kernel(const float* __restrict__ out,
                               const float* __restrict__ pre_w, const float* __restrict__ pre_b,
                               float* __restrict__ h3)
{
  long long pix = (long long)blockIdx.x * blockDim.x + threadIdx.x;
  if (pix >= (long long)BATCH * HW) return;
  const float* xr = out + pix * NF;
  int b = (int)(pix / HW), p = (int)(pix % HW);
  for (int c = 0; c < 3; ++c) {
    float a = pre_b[c];
    const float* w = pre_w + c * NF;
    for (int i = 0; i < NF; ++i) a += w[i] * xr[i];
    h3[((long long)b * 3 + c) * HW + p] = a;
  }
}

// ---------------- sobel + sin + post conv + gating ----------------
__global__ void sobel_post_kernel(const float* __restrict__ h3,
                                  const float* __restrict__ out,
                                  const float* __restrict__ post_w, const float* __restrict__ post_b,
                                  float* __restrict__ tmp)
{
  long long pix = (long long)blockIdx.x * blockDim.x + threadIdx.x;
  if (pix >= (long long)BATCH * HW) return;
  int b = (int)(pix / HW), p = (int)(pix % HW);
  int y = p / IMGD, xcol = p % IMGD;
  const float* hb = h3 + (long long)b * 3 * HW;

  const int ksz[4] = {3, 5, 9, 17};
  float sj[24];
  int jo = 0;
  for (int s = 0; s < 4; ++s) {
    int k = ksz[s], pad = (k - 1) >> 1;
    float ax0=0.f, ax1=0.f, ax2=0.f, ay0=0.f, ay1=0.f, ay2=0.f;
    for (int ky = 0; ky < k; ++ky) {
      int yy = y + ky - pad;
      if (yy < 0 || yy >= IMGD) continue;
      float ry = (float)(ky - pad);
      for (int kx = 0; kx < k; ++kx) {
        int xx = xcol + kx - pad;
        if (xx < 0 || xx >= IMGD) continue;
        float rx = (float)(kx - pad);
        float d = rx * rx + ry * ry;
        if (d == 0.f) continue;                 // center weight is zero
        float wx = rx / d, wy = ry / d;
        float v0 = hb[0 * HW + yy * IMGD + xx];
        float v1 = hb[1 * HW + yy * IMGD + xx];
        float v2 = hb[2 * HW + yy * IMGD + xx];
        ax0 += wx * v0; ay0 += wy * v0;
        ax1 += wx * v1; ay1 += wy * v1;
        ax2 += wx * v2; ay2 += wy * v2;
      }
    }
    sj[jo++] = sinf(ax0); sj[jo++] = sinf(ay0);
    sj[jo++] = sinf(ax1); sj[jo++] = sinf(ay1);
    sj[jo++] = sinf(ax2); sj[jo++] = sinf(ay2);
  }
  const float* orow = out + pix * NF;
  float* trow = tmp + pix * NF;
  for (int o = 0; o < NF; ++o) {
    float a = post_b[o];
    const float* pw = post_w + o * 24;
    for (int j = 0; j < 24; ++j) a += pw[j] * sj[j];
    trow[o] = orow[o] * a;
  }
}

// ---------------- instance-norm stats (mean, rstd per (b,c)) ---------------
__global__ void instnorm_stats_kernel(const float* __restrict__ tmp, float* __restrict__ stats)
{
  int bc = blockIdx.x;                 // b*NF + c
  int b = bc / NF, c = bc % NF;
  const float* base = tmp + (long long)b * HW * NF + c;
  float s = 0.f, q = 0.f;
  for (int p = threadIdx.x; p < HW; p += blockDim.x) {
    float v = base[(long long)p * NF];
    s += v; q += v * v;
  }
  __shared__ float shs[256], shq[256];
  shs[threadIdx.x] = s; shq[threadIdx.x] = q;
  __syncthreads();
  for (int st = 128; st > 0; st >>= 1) {
    if (threadIdx.x < st) { shs[threadIdx.x] += shs[threadIdx.x + st]; shq[threadIdx.x] += shq[threadIdx.x + st]; }
    __syncthreads();
  }
  if (threadIdx.x == 0) {
    float m = shs[0] / (float)HW;
    float var = shq[0] / (float)HW - m * m;
    stats[2 * bc]     = m;
    stats[2 * bc + 1] = rsqrtf(var + 1e-5f);
  }
}

__global__ void instnorm_apply_kernel(float* __restrict__ tmp, const float* __restrict__ stats)
{
  long long idx = (long long)blockIdx.x * blockDim.x + threadIdx.x;
  if (idx >= ACTN) return;
  int c = (int)(idx % NF);
  int b = (int)(idx / ((long long)HW * NF));
  int sidx = b * NF + c;
  tmp[idx] = (tmp[idx] - stats[2 * sidx]) * stats[2 * sidx + 1];
}

// ---------------- residual update + emb write (NCHW) ----------------
__global__ void update_out_kernel(float* __restrict__ out, const float* __restrict__ y,
                                  float* __restrict__ emb)
{
  long long idx = (long long)blockIdx.x * blockDim.x + threadIdx.x;
  if (idx >= ACTN) return;
  float v = out[idx] + 0.1f * y[idx];
  out[idx] = v;
  long long b = idx / ((long long)HW * NF);
  long long rem = idx % ((long long)HW * NF);
  long long p = rem / NF, c = rem % NF;
  emb[(b * NF + c) * HW + p] = v;
}

// ---------------- per-row L2 normalize of dyna weights ----------------
__global__ void nrm_rows_kernel(float* __restrict__ ks)
{
  int idx = blockIdx.x * blockDim.x + threadIdx.x;     // b*384 + row
  if (idx >= BATCH * 384) return;
  int b = idx / 384, rr = idx % 384;
  int off, cols;
  if      (rr < 128) { off = KIN_OFF  + rr * 64;          cols = 64;  }
  else if (rr < 256) { off = KMID_OFF + (rr - 128) * 128; cols = 128; }
  else if (rr < 320) { off = KOUT_OFF + (rr - 256) * 128; cols = 128; }
  else               { off = KSH_OFF  + (rr - 320) * 64;  cols = 64;  }
  float* w = ks + (long long)b * KS_TOTAL + off;
  float ss = 0.f;
  for (int i = 0; i < cols; ++i) ss += w[i] * w[i];
  float r = rsqrtf(ss + 1e-8f);
  for (int i = 0; i < cols; ++i) w[i] *= r;
}

// ---------------- spatial mean -> feat (B x NF) ----------------
__global__ void feat_mean_kernel(const float* __restrict__ act, float* __restrict__ feat)
{
  int bc = blockIdx.x;
  int b = bc / NF, c = bc % NF;
  const float* base = act + (long long)b * HW * NF + c;
  float s = 0.f;
  for (int p = threadIdx.x; p < HW; p += blockDim.x) s += base[(long long)p * NF];
  __shared__ float shs[256];
  shs[threadIdx.x] = s;
  __syncthreads();
  for (int st = 128; st > 0; st >>= 1) {
    if (threadIdx.x < st) shs[threadIdx.x] += shs[threadIdx.x + st];
    __syncthreads();
  }
  if (threadIdx.x == 0) feat[bc] = shs[0] / (float)HW;
}

// ---------------- host-side GEMM launchers ----------------
static inline void gemm(hipStream_t s,
                        const float* A, long long aB, int lda,
                        const float* W, long long wB, int ldw,
                        const float* bias, long long bB,
                        float* C, long long cB, int ldc,
                        int M, int N, int K, int batch, int flags)
{
  dim3 blk(32, 4, 1);
  dim3 grd((unsigned)((N / 16 + 3) / 4), (unsigned)(M / 16), (unsigned)batch);
  wmma_lin_kernel<<<grd, blk, 0, s>>>(A, aB, lda, W, wB, ldw, bias, bB, C, cB, ldc, N, K, flags);
}

static inline void gemm_big(hipStream_t s,
                            const float* A, long long aB, int lda,
                            const float* W, long long wB, int ldw,
                            const float* bias, long long bB,
                            float* C, long long cB, int ldc,
                            int M, int N, int K, int batch, int flags)
{
  dim3 blk(32, 4, 1);
  dim3 grd((unsigned)(N / 16), (unsigned)(M / 256), (unsigned)batch);
  wmma_lin_big_kernel<<<grd, blk, 0, s>>>(A, aB, lda, W, wB, ldw, bias, bB, C, cB, ldc, N, K, flags);
}

// ---------------- orchestration ----------------
extern "C" void kernel_launch(void* const* d_in, const int* in_sizes, int n_in,
                              void* d_out, int out_size, void* d_ws, size_t ws_size,
                              hipStream_t stream)
{
  (void)in_sizes; (void)n_in; (void)out_size; (void)ws_size;
  const float* x       = (const float*)d_in[0];
  const float* inj_lat = (const float*)d_in[1];
  const float* ic_w0 = (const float*)d_in[2];
  const float* ic_b0 = (const float*)d_in[3];
  const float* ic_w1 = (const float*)d_in[4];
  const float* ic_b1 = (const float*)d_in[5];
  const float* ic_ws = (const float*)d_in[6];
  const float* ic_bs = (const float*)d_in[7];
  const float* pre_w = (const float*)d_in[8];
  const float* pre_b = (const float*)d_in[9];
  const float* post_w= (const float*)d_in[10];
  const float* post_b= (const float*)d_in[11];
  const float* hy_w0 = (const float*)d_in[12];
  const float* hy_b0 = (const float*)d_in[13];
  const float* hy_w1 = (const float*)d_in[14];
  const float* hy_b1 = (const float*)d_in[15];
  const float* hy_ws = (const float*)d_in[16];
  const float* hy_bs = (const float*)d_in[17];
  const float* oc_w0 = (const float*)d_in[18];
  const float* oc_b0 = (const float*)d_in[19];
  const float* oc_w1 = (const float*)d_in[20];
  const float* oc_b1 = (const float*)d_in[21];
  const float* oc_ws = (const float*)d_in[22];
  const float* oc_bs = (const float*)d_in[23];
  const float* l0_w0 = (const float*)d_in[24];
  const float* l0_b0 = (const float*)d_in[25];
  const float* l0_w1 = (const float*)d_in[26];
  const float* l0_b1 = (const float*)d_in[27];
  const float* l0_ws = (const float*)d_in[28];
  const float* l0_bs = (const float*)d_in[29];
  const float* l1_w0 = (const float*)d_in[30];
  const float* l1_b0 = (const float*)d_in[31];
  const float* l1_w1 = (const float*)d_in[32];
  const float* l1_b1 = (const float*)d_in[33];
  const float* l1_ws = (const float*)d_in[34];
  const float* l1_bs = (const float*)d_in[35];
  const float* lf_w  = (const float*)d_in[36];
  const float* lf_b  = (const float*)d_in[37];

  float* lat_out = (float*)d_out;                   // (32,512)
  float* embs    = lat_out + BATCH * LATD;          // (9,32,64,64,64) NCHW

  float* W      = (float*)d_ws;
  float* OUT    = W;                      // 8,388,608  activation NHWC
  float* TMP    = W + 8388608LL;          // 8,388,608  out_new / out2 NHWC
  float* H3     = W + 16777216LL;         //   393,216  sobel pre NCHW
  float* H1     = W + 17170432LL;         // 16,777,216 dyna hidden1 / out_conv hidden
  float* H2     = W + 33947648LL;         // 16,777,216 dyna hidden2
  float* Y      = W + 50724864LL;         // 8,388,608  dyna output
  float* KS     = W + 59113472LL;         // 1,191,936  hyper output (B x 37248)
  float* HID    = W + 60305408LL;         //    32,768  hyper hidden (32 x 1024)
  float* STATS  = W + 60338176LL;         //     4,096  instnorm mean/rstd
  float* FEAT   = W + 60342272LL;         //     2,048  (32 x 64)
  float* LATH   = W + 60344320LL;         //    32,768  l0/l1 hidden
  float* LAT0   = W + 60377088LL;         //    16,384
  float* LAT1   = W + 60393472LL;         //    16,384

  const long long NPIXLL = (long long)BATCH * HW;

  // ---- input resblock + emb[0] ----
  {
    dim3 blk(64, 4, 1);
    dim3 grd((unsigned)(NPIXLL / 4), 1, 1);
    in_conv_kernel<<<grd, blk, 0, stream>>>(x, ic_w0, ic_b0, ic_w1, ic_b1, ic_ws, ic_bs,
                                            OUT, embs);
  }

  // ---- hypernetwork (loop-invariant: inj_lat fixed -> hoisted) ----
  gemm(stream, inj_lat, 0, LATD, hy_w0, 0, LATD, hy_b0, 0, HID, 0, 1024,
       BATCH, 1024, LATD, 1, FLAG_LRELU);
  gemm(stream, inj_lat, 0, LATD, hy_ws, 0, LATD, hy_bs, 0, KS, 0, KS_TOTAL,
       BATCH, KS_TOTAL, LATD, 1, 0);
  gemm(stream, HID, 0, 1024, hy_w1, 0, 1024, hy_b1, 0, KS, 0, KS_TOTAL,
       BATCH, KS_TOTAL, 1024, 1, FLAG_ACC);
  {
    int n = BATCH * 384;
    nrm_rows_kernel<<<(n + 255) / 256, 256, 0, stream>>>(KS);
  }

  // ---- iterated body ----
  for (int it = 0; it < NCALLS; ++it) {
    sob_pre_kernel<<<(unsigned)((NPIXLL + 255) / 256), 256, 0, stream>>>(OUT, pre_w, pre_b, H3);
    sobel_post_kernel<<<(unsigned)((NPIXLL + 255) / 256), 256, 0, stream>>>(H3, OUT, post_w, post_b, TMP);
    instnorm_stats_kernel<<<BATCH * NF, 256, 0, stream>>>(TMP, STATS);
    instnorm_apply_kernel<<<(unsigned)((ACTN + 255) / 256), 256, 0, stream>>>(TMP, STATS);

    // dyna block: per-sample 1x1 convs as batched WMMA GEMMs (M=4096 per batch)
    gemm_big(stream, TMP, (long long)HW * NF, NF,
             KS + KIN_OFF, KS_TOTAL, NF, KS + BIN_OFF, KS_TOTAL,
             H1, (long long)HW * FHID, FHID, HW, FHID, NF, BATCH, FLAG_LRELU);
    gemm_big(stream, H1, (long long)HW * FHID, FHID,
             KS + KMID_OFF, KS_TOTAL, FHID, KS + BMID_OFF, KS_TOTAL,
             H2, (long long)HW * FHID, FHID, HW, FHID, FHID, BATCH, FLAG_LRELU);
    gemm_big(stream, TMP, (long long)HW * NF, NF,
             KS + KSH_OFF, KS_TOTAL, NF, KS + BSH_OFF, KS_TOTAL,
             Y, (long long)HW * NF, NF, HW, NF, NF, BATCH, 0);
    gemm_big(stream, H2, (long long)HW * FHID, FHID,
             KS + KOUT_OFF, KS_TOTAL, FHID, KS + BOUT_OFF, KS_TOTAL,
             Y, (long long)HW * NF, NF, HW, NF, FHID, BATCH, FLAG_ACC);

    float* emb_slot = embs + (long long)(it + 1) * BATCH * NF * HW;
    update_out_kernel<<<(unsigned)((ACTN + 255) / 256), 256, 0, stream>>>(OUT, Y, emb_slot);
  }

  // ---- output resblock (shared weights, M = B*HW = 131072) ----
  const int MALL = BATCH * HW;
  gemm_big(stream, OUT, 0, NF, oc_w0, 0, NF, oc_b0, 0, H1, 0, NF,
           MALL, NF, NF, 1, FLAG_LRELU);
  gemm_big(stream, OUT, 0, NF, oc_ws, 0, NF, oc_bs, 0, TMP, 0, NF,
           MALL, NF, NF, 1, 0);
  gemm_big(stream, H1, 0, NF, oc_w1, 0, NF, oc_b1, 0, TMP, 0, NF,
           MALL, NF, NF, 1, FLAG_ACC);

  // ---- feat = mean over HW ----
  feat_mean_kernel<<<BATCH * NF, 256, 0, stream>>>(TMP, FEAT);

  // ---- latent head (M=32 path) ----
  gemm(stream, FEAT, 0, NF, l0_w0, 0, NF, l0_b0, 0, LATH, 0, 1024,
       BATCH, 1024, NF, 1, FLAG_LRELU);
  gemm(stream, FEAT, 0, NF, l0_ws, 0, NF, l0_bs, 0, LAT0, 0, LATD,
       BATCH, LATD, NF, 1, 0);
  gemm(stream, LATH, 0, 1024, l0_w1, 0, 1024, l0_b1, 0, LAT0, 0, LATD,
       BATCH, LATD, 1024, 1, FLAG_ACC);
  gemm(stream, LAT0, 0, LATD, l1_w0, 0, LATD, l1_b0, 0, LATH, 0, LATD,
       BATCH, LATD, LATD, 1, FLAG_LRELU);
  gemm(stream, LAT0, 0, LATD, l1_ws, 0, LATD, l1_bs, 0, LAT1, 0, LATD,
       BATCH, LATD, LATD, 1, 0);
  gemm(stream, LATH, 0, LATD, l1_w1, 0, LATD, l1_b1, 0, LAT1, 0, LATD,
       BATCH, LATD, LATD, 1, FLAG_ACC);
  gemm(stream, LAT1, 0, LATD, lf_w, 0, LATD, lf_b, 0, lat_out, 0, LATD,
       BATCH, LATD, LATD, 1, 0);
}
